// Wide_BasicBlock_Q_35356170781052
// MI455X (gfx1250) — 8-bit matrix operand order repaired and checked by reading, compiled, not run on hardware
//
#include <hip/hip_runtime.h>
#include <hip/hip_bf16.h>


typedef int v8i __attribute__((ext_vector_type(8)));

#define NB   256
#define CH   256
#define HW   16
#define EPSV 1e-5

__global__ void init_stats_kernel(unsigned int* mx1, unsigned int* mx2,
                                  unsigned long long* sum2,
                                  unsigned long long* sq2) {
  int t = threadIdx.x;
  sum2[t] = 0ull;
  sq2[t]  = 0ull;
  if (t == 0) { *mx1 = 0u; *mx2 = 0u; }
}

__global__ void bn1_stats_kernel(const float* __restrict__ x,
                                 float* __restrict__ mean1,
                                 float* __restrict__ inv1) {
  __shared__ double sred[256];
  __shared__ double qred[256];
  int c = blockIdx.x, t = threadIdx.x;
  double s = 0.0, q = 0.0;
  for (int n = 0; n < NB; ++n) {
    float v = x[((size_t)(n * CH + c) << 8) + t];
    s += v;
    q += (double)v * v;
  }
  sred[t] = s; qred[t] = q;
  __syncthreads();
  for (int st = 128; st > 0; st >>= 1) {
    if (t < st) { sred[t] += sred[t + st]; qred[t] += qred[t + st]; }
    __syncthreads();
  }
  if (t == 0) {
    double m   = sred[0] / 65536.0;
    double var = qred[0] / 65536.0 - m * m;
    mean1[c] = (float)m;
    inv1[c]  = (float)(1.0 / sqrt(var + EPSV));
  }
}

__global__ void act1_kernel(const float* __restrict__ x,
                            const float* __restrict__ mean,
                            const float* __restrict__ inv,
                            const float* __restrict__ gamma,
                            const float* __restrict__ beta,
                            signed char* __restrict__ a) {
  unsigned idx = blockIdx.x * blockDim.x + threadIdx.x;
  unsigned wq = idx & 15u;
  unsigned hq = (idx >> 4) & 15u;
  unsigned c  = (idx >> 8) & 255u;
  unsigned n  = idx >> 16;
  float v = (x[idx] - mean[c]) * inv[c] * gamma[c] + beta[c];
  v = fminf(fmaxf(v, 0.f), 1.f);
  int q = (int)rintf(v * 15.f);
  unsigned spat = (n << 8) + (hq << 4) + wq;
  a[((size_t)spat << 8) + c] = (signed char)q;
}

__global__ void wmax_kernel(const float* __restrict__ w, int n,
                            unsigned int* __restrict__ outbits) {
  __shared__ float red[256];
  float m = 0.f;
  for (int i = blockIdx.x * blockDim.x + threadIdx.x; i < n;
       i += gridDim.x * blockDim.x)
    m = fmaxf(m, fabsf(tanhf(w[i])));
  red[threadIdx.x] = m;
  __syncthreads();
  for (int s = 128; s > 0; s >>= 1) {
    if (threadIdx.x < s)
      red[threadIdx.x] = fmaxf(red[threadIdx.x], red[threadIdx.x + s]);
    __syncthreads();
  }
  if (threadIdx.x == 0) atomicMax(outbits, __float_as_uint(red[0]));
}

__global__ void wquant_kernel(const float* __restrict__ w,
                              const int* __restrict__ pat,
                              const unsigned int* __restrict__ mxbits,
                              signed char* __restrict__ qw) {
  const int KH[5] = {0, 1, 1, 1, 2};
  const int KW[5] = {1, 0, 1, 2, 1};
  unsigned idx = blockIdx.x * blockDim.x + threadIdx.x;
  if (idx >= 5u * 256u * 256u) return;
  unsigned ic = idx & 255u, oc = (idx >> 8) & 255u, t = idx >> 16;
  float mx = __uint_as_float(*mxbits);
  float v  = w[(size_t)(oc * 256 + ic) * 9 + KH[t] * 3 + KW[t]];
  int mask = pat[KH[t] * 3 + KW[t]];
  int q = 2 * (int)rintf((tanhf(v) / (2.f * mx) + 0.5f) * 15.f) - 15;
  qw[idx] = (signed char)(mask ? q : 0);
}

__device__ __forceinline__ v8i conv_tile(const signed char* __restrict__ act,
                                         const signed char* __restrict__ qw,
                                         int n, int h, int ocb,
                                         int low, int half) {
  const int dH[5] = {-1, 0, 0, 0, 1};
  const int dW[5] = {0, -1, 0, 1, 0};
  v8i acc = {0, 0, 0, 0, 0, 0, 0, 0};
#pragma unroll
  for (int t = 0; t < 5; ++t) {
    int hh = h + dH[t];
    int ww = low + dW[t];
    bool ok = ((unsigned)hh < 16u) && ((unsigned)ww < 16u);
    long aoff = ((((long)n * 16 + hh) * 16 + ww) << 8) + (half << 4);
    const signed char* wrow =
        qw + (((long)t * 256 + ocb * 16 + low) << 8) + (half << 4);
#pragma unroll
    for (int kb = 0; kb < 4; ++kb) {
      v8i A = {0, 0, 0, 0, 0, 0, 0, 0};
      if (ok) {
        const uint2* pa = reinterpret_cast<const uint2*>(act + aoff + kb * 64);
        uint2 a0 = pa[0], a1 = pa[1], a2 = pa[4], a3 = pa[5];
        A[0] = (int)a0.x; A[1] = (int)a0.y;
        A[2] = (int)a1.x; A[3] = (int)a1.y;
        A[4] = (int)a2.x; A[5] = (int)a2.y;
        A[6] = (int)a3.x; A[7] = (int)a3.y;
      }
      const uint4* pb = reinterpret_cast<const uint4*>(wrow + kb * 64);
      uint4 b0 = pb[0], b1 = pb[2];
      v8i B = {(int)b0.x, (int)b0.y, (int)b0.z, (int)b0.w,
               (int)b1.x, (int)b1.y, (int)b1.z, (int)b1.w};
      acc = __builtin_amdgcn_wmma_i32_16x16x64_iu8(false, A, true, B, acc,
                                                   false, false);
    }
  }
  return acc;
}

__device__ __forceinline__ long long shfl_xor_ll(long long v, int mask) {
  int lo = (int)(v & 0xffffffffll);
  int hi = (int)(v >> 32);
  lo = __shfl_xor(lo, mask, 32);
  hi = __shfl_xor(hi, mask, 32);
  return ((long long)hi << 32) | (unsigned int)lo;
}

__global__ void conv1_kernel(const signed char* __restrict__ act,
                             const signed char* __restrict__ qw,
                             int* __restrict__ y1,
                             unsigned long long* __restrict__ sum2,
                             unsigned long long* __restrict__ sq2) {
  int wave = threadIdx.x >> 5, lane = threadIdx.x & 31;
  int tile = blockIdx.x * 8 + wave;
  int ocb = tile & 15, nh = tile >> 4;
  int h = nh & 15, n = nh >> 4;
  int low = lane & 15, half = lane >> 4;

  v8i acc = conv_tile(act, qw, n, h, ocb, low, half);

  long long s = 0, q = 0;
#pragma unroll
  for (int r = 0; r < 8; ++r) {
    int m = r + half * 8;
    int v = acc[r];
    y1[((size_t)(nh * 16 + m) << 8) + ocb * 16 + low] = v;
    s += v;
    q += (long long)v * v;
  }
  s += shfl_xor_ll(s, 16);
  q += shfl_xor_ll(q, 16);
  if (lane < 16) {
    atomicAdd(&sum2[ocb * 16 + lane], (unsigned long long)s);
    atomicAdd(&sq2[ocb * 16 + lane], (unsigned long long)q);
  }
}

__global__ void bn2_finalize_kernel(const unsigned long long* __restrict__ sum2,
                                    const unsigned long long* __restrict__ sq2,
                                    float* __restrict__ mean2,
                                    float* __restrict__ inv2) {
  int c = threadIdx.x;
  double s = (double)(long long)sum2[c];
  double q = (double)sq2[c];
  double m  = s / (65536.0 * 225.0);
  double e2 = q / (65536.0 * 225.0 * 225.0);
  double var = e2 - m * m;
  mean2[c] = (float)m;
  inv2[c]  = (float)(1.0 / sqrt(var + EPSV));
}

__global__ void act2_kernel(const int* __restrict__ y1,
                            const float* __restrict__ mean,
                            const float* __restrict__ inv,
                            const float* __restrict__ gamma,
                            const float* __restrict__ beta,
                            signed char* __restrict__ a) {
  unsigned idx = blockIdx.x * blockDim.x + threadIdx.x;
  unsigned c = idx & 255u;
  float v = (float)y1[idx] * (1.f / 225.f);
  v = (v - mean[c]) * inv[c] * gamma[c] + beta[c];
  v = fminf(fmaxf(v, 0.f), 1.f);
  a[idx] = (signed char)(int)rintf(v * 15.f);
}

__global__ void conv2_kernel(const signed char* __restrict__ act,
                             const signed char* __restrict__ qw,
                             const float* __restrict__ x,
                             float* __restrict__ out) {
  __shared__ int lds[8][16 * 17];
  int wave = threadIdx.x >> 5, lane = threadIdx.x & 31;
  int tile = blockIdx.x * 8 + wave;
  int ocb = tile & 15, nh = tile >> 4;
  int h = nh & 15, n = nh >> 4;
  int low = lane & 15, half = lane >> 4;

  v8i acc = conv_tile(act, qw, n, h, ocb, low, half);

  int* my = lds[wave];
#pragma unroll
  for (int r = 0; r < 8; ++r) my[(r + half * 8) * 17 + low] = acc[r];
  __syncthreads();
#pragma unroll
  for (int o2 = 0; o2 < 8; ++o2) {
    int oc_l = o2 * 2 + half;
    float v = (float)my[low * 17 + oc_l] * (1.f / 225.f);
    size_t oi = (((size_t)(n * 256 + ocb * 16 + oc_l)) * 16 + h) * 16 + low;
    out[oi] = v + x[oi];
  }
}

extern "C" void kernel_launch(void* const* d_in, const int* in_sizes, int n_in,
                              void* d_out, int out_size, void* d_ws,
                              size_t ws_size, hipStream_t stream) {
  const float* x  = (const float*)d_in[0];
  const float* g1 = (const float*)d_in[1];
  const float* b1 = (const float*)d_in[2];
  const float* w1 = (const float*)d_in[3];
  const float* g2 = (const float*)d_in[4];
  const float* b2 = (const float*)d_in[5];
  const float* w2 = (const float*)d_in[6];
  const int*   pat = (const int*)d_in[7];
  float* out = (float*)d_out;

  char* ws = (char*)d_ws;
  float* mean1 = (float*)(ws + 0);
  float* inv1  = (float*)(ws + 1024);
  float* mean2 = (float*)(ws + 2048);
  float* inv2  = (float*)(ws + 3072);
  unsigned int* mx1 = (unsigned int*)(ws + 4096);
  unsigned int* mx2 = (unsigned int*)(ws + 4160);
  unsigned long long* sum2 = (unsigned long long*)(ws + 4352);
  unsigned long long* sq2  = (unsigned long long*)(ws + 6400);
  signed char* a_buf = (signed char*)(ws + 16384);
  signed char* qw1 = (signed char*)(ws + 16384 + (1u << 24));
  signed char* qw2 = qw1 + 5 * 256 * 256;
  int* y1 = (int*)(ws + 17448960);

  const int NW = 589824;
  init_stats_kernel<<<1, 256, 0, stream>>>(mx1, mx2, sum2, sq2);
  bn1_stats_kernel<<<256, 256, 0, stream>>>(x, mean1, inv1);
  act1_kernel<<<65536, 256, 0, stream>>>(x, mean1, inv1, g1, b1, a_buf);
  wmax_kernel<<<256, 256, 0, stream>>>(w1, NW, mx1);
  wmax_kernel<<<256, 256, 0, stream>>>(w2, NW, mx2);
  wquant_kernel<<<1280, 256, 0, stream>>>(w1, pat, mx1, qw1);
  wquant_kernel<<<1280, 256, 0, stream>>>(w2, pat, mx2, qw2);
  conv1_kernel<<<8192, 256, 0, stream>>>(a_buf, qw1, y1, sum2, sq2);
  bn2_finalize_kernel<<<1, 256, 0, stream>>>(sum2, sq2, mean2, inv2);
  act2_kernel<<<65536, 256, 0, stream>>>(y1, mean2, inv2, g2, b2, a_buf);
  conv2_kernel<<<8192, 256, 0, stream>>>(a_buf, qw2, x, out);
}
